// PerformerLM_62225486184687
// MI455X (gfx1250) — matrix operand order repaired and checked by reading, not compiled, not run on hardware
//
#include <hip/hip_runtime.h>
#include <cstdint>
#include <cstddef>

// ---------------------------------------------------------------------------
// Performer LM forward (FAVOR+), CDNA5 / gfx1250, bf16 WMMA GEMM core.
// B=1, N=4096 tokens, DIM=768, HEADS=12, DH=64, FF=3072, Mfeat=256, DEPTH=6.
// ---------------------------------------------------------------------------

#define NTOK   4096
#define DIM    768
#define DEPTH  6
#define HEADS  12
#define DH     64
#define INNER  768
#define FFD    3072
#define MFEAT  256
#define NOUT   102

typedef __attribute__((ext_vector_type(16))) __bf16 v16bf;
typedef __attribute__((ext_vector_type(8)))  float  v8f;

// float -> bf16 bits, round-to-nearest-even
__device__ __forceinline__ uint32_t f2bf(float x) {
    uint32_t u = __float_as_uint(x);
    u += 0x7FFFu + ((u >> 16) & 1u);
    return u >> 16;
}
// pack two floats -> packed bf16x2
__device__ __forceinline__ uint32_t f2bf2(float lo, float hi) {
    return f2bf(lo) | (f2bf(hi) << 16);
}

__device__ __forceinline__ float gelu_tanh(float x) {
    float x3 = x * x * x;
    return 0.5f * x * (1.0f + tanhf(0.7978845608028654f * (x + 0.044715f * x3)));
}

// ---------------------------------------------------------------------------
// Generic strided, head-batched GEMM:  C = act(A*a_scale @ B + bias) * rowscale + res
// A[m,k] = A[b*a_bs + m*a_rs + k*a_cs], B[k,n], C[m,n] analogous.
// CONTRACT (enforced by the launches below): M % 128 == 0, K % 32 == 0.
// Only N may be ragged (final head, N=102) -> BMODE 2.
// AMODE: 0 = k-contig (a_cs==1, a_rs%4==0), 1 = m-contig (a_rs==1)
// BMODE: 0 = k-contig (b_rs==1, b_cs%4==0), 1 = n-contig (b_cs==1, b_rs%4==0),
//        2 = guarded scalar (ragged / unaligned N)
// ---------------------------------------------------------------------------
struct GemmArgs {
    const float* A; long a_rs, a_cs, a_bs; float a_scale;
    const float* B; long b_rs, b_cs, b_bs;
    float*       C; long c_rs, c_cs, c_bs;
    const float* bias;     long bias_bs;           // per-n bias (nullable)
    const float* res;      long r_rs, r_cs, r_bs;  // residual add (nullable)
    const float* rowscale; long rs_bs;             // per-m scale (nullable)
    int M, N, K, act;                              // act: 0 none, 1 gelu(tanh)
};

#define BM 128
#define BN 64
#define BK 32
#define LDP 40   // LDS row pitch in halves (80B) -> conflict-free b128 reads

union Frag { v16bf v; uint4 q[2]; };

// Staging: explicit load phase (all global_load_b128 in flight) then
// convert+store phase, so waits are watermarked instead of per-load.
template<int AMODE, int BMODE>
__device__ __forceinline__ void stage_tile(const GemmArgs& g,
                                           const float* __restrict__ A,
                                           const float* __restrict__ B,
                                           int row0, int col0, int k0,
                                           uint16_t* __restrict__ as,
                                           uint16_t* __restrict__ bs, int tid) {
    // ---- A tile 128x32 -> LDS row-major [m][k] ----
    if constexpr (AMODE == 0) {
        float4 fa[4];
#pragma unroll
        for (int it = 0; it < 4; ++it) {
            int t = tid + it * 256;
            int r = t >> 3, c4 = (t & 7) * 4;
            fa[it] = *reinterpret_cast<const float4*>(
                A + (long)(row0 + r) * g.a_rs + (k0 + c4));
        }
#pragma unroll
        for (int it = 0; it < 4; ++it) {
            int t = tid + it * 256;
            int r = t >> 3, c4 = (t & 7) * 4;
            uint2 p; p.x = f2bf2(fa[it].x * g.a_scale, fa[it].y * g.a_scale);
                     p.y = f2bf2(fa[it].z * g.a_scale, fa[it].w * g.a_scale);
            *reinterpret_cast<uint2*>(&as[r * LDP + c4]) = p;
        }
    } else {  // AMODE == 1: m-contiguous
        float4 fa[4];
#pragma unroll
        for (int it = 0; it < 4; ++it) {
            int t = tid + it * 256;
            int c = t >> 5, m4 = (t & 31) * 4;
            fa[it] = *reinterpret_cast<const float4*>(
                A + (long)(k0 + c) * g.a_cs + (row0 + m4));
        }
#pragma unroll
        for (int it = 0; it < 4; ++it) {
            int t = tid + it * 256;
            int c = t >> 5, m4 = (t & 31) * 4;
            as[(m4 + 0) * LDP + c] = (uint16_t)f2bf(fa[it].x * g.a_scale);
            as[(m4 + 1) * LDP + c] = (uint16_t)f2bf(fa[it].y * g.a_scale);
            as[(m4 + 2) * LDP + c] = (uint16_t)f2bf(fa[it].z * g.a_scale);
            as[(m4 + 3) * LDP + c] = (uint16_t)f2bf(fa[it].w * g.a_scale);
        }
    }
    // ---- B tile 32x64 -> LDS n-major [n][k] ----
    if constexpr (BMODE == 0) {
        float4 fb[2];
#pragma unroll
        for (int it = 0; it < 2; ++it) {
            int t = tid + it * 256;
            int n = t >> 3, c4 = (t & 7) * 4;
            fb[it] = *reinterpret_cast<const float4*>(
                B + (long)(col0 + n) * g.b_cs + (k0 + c4));
        }
#pragma unroll
        for (int it = 0; it < 2; ++it) {
            int t = tid + it * 256;
            int n = t >> 3, c4 = (t & 7) * 4;
            uint2 p; p.x = f2bf2(fb[it].x, fb[it].y); p.y = f2bf2(fb[it].z, fb[it].w);
            *reinterpret_cast<uint2*>(&bs[n * LDP + c4]) = p;
        }
    } else if constexpr (BMODE == 1) {
        float4 fb[2];
#pragma unroll
        for (int it = 0; it < 2; ++it) {
            int t = tid + it * 256;
            int c = t >> 4, n4 = (t & 15) * 4;
            fb[it] = *reinterpret_cast<const float4*>(
                B + (long)(k0 + c) * g.b_rs + (col0 + n4));
        }
#pragma unroll
        for (int it = 0; it < 2; ++it) {
            int t = tid + it * 256;
            int c = t >> 4, n4 = (t & 15) * 4;
            bs[(n4 + 0) * LDP + c] = (uint16_t)f2bf(fb[it].x);
            bs[(n4 + 1) * LDP + c] = (uint16_t)f2bf(fb[it].y);
            bs[(n4 + 2) * LDP + c] = (uint16_t)f2bf(fb[it].z);
            bs[(n4 + 3) * LDP + c] = (uint16_t)f2bf(fb[it].w);
        }
    } else {  // BMODE == 2: guarded scalar (final head only)
        float fb[8];
#pragma unroll
        for (int it = 0; it < 8; ++it) {
            int t = tid + it * 256;
            int n = t >> 5, c = t & 31;
            int gn = col0 + n;
            fb[it] = (gn < g.N)
                ? B[(long)(k0 + c) * g.b_rs + (long)gn * g.b_cs] : 0.f;
        }
#pragma unroll
        for (int it = 0; it < 8; ++it) {
            int t = tid + it * 256;
            int n = t >> 5, c = t & 31;
            bs[n * LDP + c] = (uint16_t)f2bf(fb[it]);
        }
    }
}

template<int AMODE, int BMODE>
__global__ __launch_bounds__(256) void gemm_wmma_bf16(GemmArgs g) {
    __shared__ uint16_t As[2][BM * LDP];  // double-buffered
    __shared__ uint16_t Bs[2][BN * LDP];

    const int bz   = blockIdx.z;
    const int row0 = blockIdx.y * BM;
    const int col0 = blockIdx.x * BN;
    const float* A = g.A + (long)bz * g.a_bs;
    const float* B = g.B + (long)bz * g.b_bs;

    const int tid  = threadIdx.x;
    const int wave = tid >> 5;
    const int lane = tid & 31;
    const int wn   = wave & 1;      // 2 n-subtiles of 32
    const int wm   = wave >> 1;     // 4 m-subtiles of 32
    const int mrow = lane & 15;
    const int hi   = lane >> 4;     // lane half selects K sub-block
    const int klo  = hi ? 8 : 0;

    v8f acc[2][2];
    for (int i = 0; i < 2; ++i)
        for (int j = 0; j < 2; ++j)
            acc[i][j] = (v8f){0.f,0.f,0.f,0.f,0.f,0.f,0.f,0.f};

    const int nk = g.K / BK;
    stage_tile<AMODE, BMODE>(g, A, B, row0, col0, 0, As[0], Bs[0], tid);

    for (int kt = 0; kt < nk; ++kt) {
        __syncthreads();   // buffer kt&1 ready; previous reads of (kt+1)&1 drained

        if (kt + 1 < nk)   // stage next tile into other buffer while computing
            stage_tile<AMODE, BMODE>(g, A, B, row0, col0, (kt + 1) * BK,
                                     As[(kt + 1) & 1], Bs[(kt + 1) & 1], tid);
        if (kt + 2 < nk) { // L2 prefetch two tiles ahead (global_prefetch_b8)
            int kp = (kt + 2) * BK;
            __builtin_prefetch(A + (long)(row0 + (tid >> 1)) * g.a_rs +
                               (long)(kp + (tid & 1) * 16) * g.a_cs, 0, 1);
            __builtin_prefetch(B + (long)(kp + (tid & 31)) * g.b_rs +
                               (long)(col0 + (tid >> 5) * 8) * g.b_cs, 0, 1);
        }

        const uint16_t* as = As[kt & 1];
        const uint16_t* bs = Bs[kt & 1];

        // ---- gather fragments per ISA 16-bit A/B VGPR layouts ----
        Frag a0, a1, b0, b1;
        {
            const uint16_t* p = &as[(wm * 32 + mrow) * LDP];
            a0.q[0] = *reinterpret_cast<const uint4*>(p + klo);
            a0.q[1] = *reinterpret_cast<const uint4*>(p + 16 + klo);
            p = &as[(wm * 32 + 16 + mrow) * LDP];
            a1.q[0] = *reinterpret_cast<const uint4*>(p + klo);
            a1.q[1] = *reinterpret_cast<const uint4*>(p + 16 + klo);
        }
        {
            const uint16_t* p = &bs[(wn * 32 + mrow) * LDP + klo];
            b0.q[0] = *reinterpret_cast<const uint4*>(p);
            b0.q[1] = *reinterpret_cast<const uint4*>(p + 16);
            p = &bs[(wn * 32 + 16 + mrow) * LDP + klo];
            b1.q[0] = *reinterpret_cast<const uint4*>(p);
            b1.q[1] = *reinterpret_cast<const uint4*>(p + 16);
        }

        acc[0][0] = __builtin_amdgcn_wmma_f32_16x16x32_bf16(
            false, a0.v, false, b0.v, (short)0, acc[0][0], false, false);
        acc[0][1] = __builtin_amdgcn_wmma_f32_16x16x32_bf16(
            false, a0.v, false, b1.v, (short)0, acc[0][1], false, false);
        acc[1][0] = __builtin_amdgcn_wmma_f32_16x16x32_bf16(
            false, a1.v, false, b0.v, (short)0, acc[1][0], false, false);
        acc[1][1] = __builtin_amdgcn_wmma_f32_16x16x32_bf16(
            false, a1.v, false, b1.v, (short)0, acc[1][1], false, false);
    }

    // ---- epilogue: C layout lane = col, vgpr i = row (i + 8*hi) ----
    float* C = g.C + (long)bz * g.c_bs;
    int   nn[2];
    float bias_v[2];
    for (int sn = 0; sn < 2; ++sn) {
        nn[sn] = col0 + wn * 32 + sn * 16 + mrow;
        bias_v[sn] = (g.bias && nn[sn] < g.N)
                   ? g.bias[(long)bz * g.bias_bs + nn[sn]] : 0.f;
    }
    for (int sm = 0; sm < 2; ++sm) {
        int mb = row0 + wm * 32 + sm * 16 + hi * 8;
        for (int sn = 0; sn < 2; ++sn) {
            if (nn[sn] >= g.N) continue;
            v8f a = acc[sm][sn];
            for (int i = 0; i < 8; ++i) {
                int m = mb + i;
                float v = a[i] + bias_v[sn];
                if (g.act == 1) v = gelu_tanh(v);
                if (g.rowscale) v *= g.rowscale[(long)bz * g.rs_bs + m];
                if (g.res) v += g.res[(long)bz * g.r_bs + (long)m * g.r_rs +
                                      (long)nn[sn] * g.r_cs];
                C[(long)m * g.c_rs + (long)nn[sn] * g.c_cs] = v;
            }
        }
    }
}

// ---------------------------------------------------------------------------
// Elementwise / reduction kernels
// ---------------------------------------------------------------------------

// bucketize(methy) + 3-table gather; also writes mask
__global__ __launch_bounds__(256) void embed_kernel(
    const float* __restrict__ methy, const int* __restrict__ chromo,
    const int* __restrict__ pos, const float* __restrict__ mt,
    const float* __restrict__ ct, const float* __restrict__ pt,
    float* __restrict__ x, float* __restrict__ mask) {
    int tok = blockIdx.x;
    __shared__ int sidx;
    float v = methy[tok];
    if (threadIdx.x == 0) {
        // boundaries[:-1] = [-2, -1, 0.00, 0.01, ..., 0.99]; searchsorted 'left'
        int idx = 0;
        if (v > -2.f) idx++;
        if (v > -1.f) idx++;
        for (int j = 0; j < 100; ++j) if (v > (float)j * 0.01f) idx++;
        sidx = idx;
        mask[tok] = (v != 0.f) ? 1.f : 0.f;
    }
    __syncthreads();
    int idx = sidx, c = chromo[tok], p = pos[tok];
    for (int d = threadIdx.x; d < DIM; d += 256)
        x[(long)tok * DIM + d] =
            mt[(long)idx * DIM + d] + pt[(long)p * DIM + d] + ct[(long)c * DIM + d];
}

__global__ __launch_bounds__(256) void ln_kernel(
    const float* __restrict__ x, const float* __restrict__ gam,
    const float* __restrict__ bet, float* __restrict__ y) {
    int row = blockIdx.x;
    const float* xr = x + (long)row * DIM;
    float* yr = y + (long)row * DIM;
    __shared__ float s1[256], s2[256];
    float a = 0.f, c = 0.f;
    for (int d = threadIdx.x; d < DIM; d += 256) { float v = xr[d]; a += v; c += v * v; }
    s1[threadIdx.x] = a; s2[threadIdx.x] = c; __syncthreads();
    for (int off = 128; off > 0; off >>= 1) {
        if (threadIdx.x < off) { s1[threadIdx.x] += s1[threadIdx.x + off];
                                 s2[threadIdx.x] += s2[threadIdx.x + off]; }
        __syncthreads();
    }
    float mu = s1[0] * (1.f / DIM);
    float var = s2[0] * (1.f / DIM) - mu * mu;
    float r = rsqrtf(var + 1e-5f);
    for (int d = threadIdx.x; d < DIM; d += 256)
        yr[d] = (xr[d] - mu) * r * gam[d] + bet[d];
}

// diag[h*N+n] = 0.5*dnorm^2*sum_d q[n, h*64+d]^2 = sum(q^2)/16  (dnorm=64^-0.25)
__global__ void diag_kernel(const float* __restrict__ qk, float* __restrict__ out) {
    int i = blockIdx.x * blockDim.x + threadIdx.x;
    if (i >= HEADS * NTOK) return;
    int h = i / NTOK, n = i % NTOK;
    const float* p = qk + (long)n * INNER + h * DH;
    float s = 0.f;
    for (int d = 0; d < DH; ++d) s += p[d] * p[d];
    out[i] = s * (1.0f / 16.0f);
}

__global__ void rowmax_kernel(const float* __restrict__ xd, float* __restrict__ out) {
    int i = blockIdx.x * blockDim.x + threadIdx.x;
    if (i >= HEADS * NTOK) return;
    const float* r = xd + (long)i * MFEAT;
    float m = r[0];
    for (int j = 1; j < MFEAT; ++j) m = fmaxf(m, r[j]);
    out[i] = m;
}

__global__ __launch_bounds__(256) void headmax_kernel(
    const float* __restrict__ rows, float* __restrict__ out) {
    int h = blockIdx.x;
    __shared__ float s[256];
    float m = -3.0e38f;
    for (int n = threadIdx.x; n < NTOK; n += 256) m = fmaxf(m, rows[h * NTOK + n]);
    s[threadIdx.x] = m; __syncthreads();
    for (int off = 128; off > 0; off >>= 1) {
        if (threadIdx.x < off) s[threadIdx.x] = fmaxf(s[threadIdx.x], s[threadIdx.x + off]);
        __syncthreads();
    }
    if (threadIdx.x == 0) out[h] = s[0];
}

// qf = ratio*(exp(xd - diag - rowmax) + eps), in place.  ratio=256^-0.5
__global__ void expq_kernel(float* __restrict__ xd, const float* __restrict__ diag,
                            const float* __restrict__ mx) {
    long i = (long)blockIdx.x * 256 + threadIdx.x;
    long row = i >> 8;
    xd[i] = 0.0625f * (expf(xd[i] - diag[row] - mx[row]) + 1e-4f);
}

// kf = ratio*(exp(xd - diag - headmax) + eps) * mask[n], in place
__global__ void expk_kernel(float* __restrict__ xd, const float* __restrict__ diag,
                            const float* __restrict__ mxh, const float* __restrict__ mask) {
    long i = (long)blockIdx.x * 256 + threadIdx.x;
    long row = i >> 8;
    int h = (int)(row >> 12);
    int n = (int)(row & (NTOK - 1));
    xd[i] = 0.0625f * (expf(xd[i] - diag[row] - mxh[h]) + 1e-4f) * mask[n];
}

// k_sum[h][m] = sum_n kf[h][n][m]
__global__ __launch_bounds__(256) void colsum_kernel(
    const float* __restrict__ kf, float* __restrict__ ksum) {
    int h = blockIdx.x, m = threadIdx.x;
    const float* base = kf + (long)h * NTOK * MFEAT;
    float s = 0.f;
    for (int n = 0; n < NTOK; ++n) s += base[(long)n * MFEAT + m];
    ksum[h * MFEAT + m] = s;
}

// d_inv[h*N+n] = 1 / dot(qf[h,n,:], k_sum[h,:])
__global__ void dinv_kernel(const float* __restrict__ qf, const float* __restrict__ ksum,
                            float* __restrict__ dinv) {
    int i = blockIdx.x * blockDim.x + threadIdx.x;
    if (i >= HEADS * NTOK) return;
    int h = i >> 12;
    const float* r = qf + (long)i * MFEAT;
    const float* s = ksum + h * MFEAT;
    float d = 0.f;
    for (int j = 0; j < MFEAT; ++j) d += r[j] * s[j];
    dinv[i] = 1.0f / d;
}

// ---------------------------------------------------------------------------
// Host side
// ---------------------------------------------------------------------------
static void launch_gemm(hipStream_t st, int amode, int bmode,
                        const float* A, long ars, long acs, long abs_, float ascale,
                        const float* B, long brs, long bcs, long bbs,
                        float* C, long crs, long ccs, long cbs,
                        const float* bias, long biasbs,
                        const float* res, long rrs, long rcs, long rbs,
                        const float* rowscale, long rsbs,
                        int M, int N, int K, int act, int batch) {
    GemmArgs g;
    g.A = A; g.a_rs = ars; g.a_cs = acs; g.a_bs = abs_; g.a_scale = ascale;
    g.B = B; g.b_rs = brs; g.b_cs = bcs; g.b_bs = bbs;
    g.C = C; g.c_rs = crs; g.c_cs = ccs; g.c_bs = cbs;
    g.bias = bias; g.bias_bs = biasbs;
    g.res = res; g.r_rs = rrs; g.r_cs = rcs; g.r_bs = rbs;
    g.rowscale = rowscale; g.rs_bs = rsbs;
    g.M = M; g.N = N; g.K = K; g.act = act;
    dim3 grid((N + BN - 1) / BN, (M + BM - 1) / BM, batch);
    if (amode == 0 && bmode == 0)      gemm_wmma_bf16<0, 0><<<grid, 256, 0, st>>>(g);
    else if (amode == 0 && bmode == 1) gemm_wmma_bf16<0, 1><<<grid, 256, 0, st>>>(g);
    else if (amode == 1 && bmode == 1) gemm_wmma_bf16<1, 1><<<grid, 256, 0, st>>>(g);
    else                               gemm_wmma_bf16<0, 2><<<grid, 256, 0, st>>>(g);
}

extern "C" void kernel_launch(void* const* d_in, const int* in_sizes, int n_in,
                              void* d_out, int out_size, void* d_ws, size_t ws_size,
                              hipStream_t stream) {
    (void)in_sizes; (void)n_in; (void)out_size; (void)ws_size;

    const float* methy        = (const float*)d_in[0];
    const int*   chromo       = (const int*)d_in[1];
    const int*   pos          = (const int*)d_in[2];
    const float* methy_table  = (const float*)d_in[3];
    const float* chromo_table = (const float*)d_in[4];
    const float* pos_table    = (const float*)d_in[5];
    const float* ln1_g = (const float*)d_in[6];
    const float* ln1_b = (const float*)d_in[7];
    const float* ln2_g = (const float*)d_in[8];
    const float* ln2_b = (const float*)d_in[9];
    const float* Wq = (const float*)d_in[10];
    const float* Wk = (const float*)d_in[11];
    const float* Wv = (const float*)d_in[12];
    const float* Wo = (const float*)d_in[13];
    const float* bo = (const float*)d_in[14];
    const float* W1 = (const float*)d_in[15];
    const float* b1 = (const float*)d_in[16];
    const float* W2 = (const float*)d_in[17];
    const float* b2 = (const float*)d_in[18];
    const float* proj = (const float*)d_in[19];
    const float* normf_g = (const float*)d_in[20];
    const float* normf_b = (const float*)d_in[21];
    const float* Wout = (const float*)d_in[22];
    const float* bout = (const float*)d_in[23];

    // workspace carve-out (~178 MB)
    const size_t SZ_X = (size_t)NTOK * DIM * 4;              // 12.6 MB
    const size_t SZ_F = (size_t)HEADS * NTOK * MFEAT * 4;    // 50.3 MB (== NTOK*FFD*4)
    char* base = (char*)d_ws;
    float* x    = (float*)(base);
    float* h    = (float*)(base + SZ_X);
    float* q    = (float*)(base + 2 * SZ_X);
    float* k    = (float*)(base + 3 * SZ_X);
    float* v    = (float*)(base + 4 * SZ_X);
    float* o    = (float*)(base + 5 * SZ_X);
    float* qf   = (float*)(base + 6 * SZ_X);                 // xd_q / qf
    float* kf   = (float*)(base + 6 * SZ_X + SZ_F);          // xd_k / kf / FF scratch
    float* ctx  = (float*)(base + 6 * SZ_X + 2 * SZ_F);      // [12,256,64]
    char*  sm   = base + 6 * SZ_X + 2 * SZ_F + (size_t)HEADS * MFEAT * DH * 4;
    float* diagq  = (float*)(sm);                 sm += (size_t)HEADS * NTOK * 4;
    float* diagk  = (float*)(sm);                 sm += (size_t)HEADS * NTOK * 4;
    float* mxq    = (float*)(sm);                 sm += (size_t)HEADS * NTOK * 4;
    float* mxrows = (float*)(sm);                 sm += (size_t)HEADS * NTOK * 4;
    float* mxk    = (float*)(sm);                 sm += 256;
    float* ksum   = (float*)(sm);                 sm += (size_t)HEADS * MFEAT * 4;
    float* dinv   = (float*)(sm);                 sm += (size_t)HEADS * NTOK * 4;
    float* mask   = (float*)(sm);

    const int HN = HEADS * NTOK;                 // 49152
    const float dnorm = 0.35355339059327379f;    // 64^-0.25

    embed_kernel<<<NTOK, 256, 0, stream>>>(methy, chromo, pos, methy_table,
                                           chromo_table, pos_table, x, mask);

    for (int l = 0; l < DEPTH; ++l) {
        const float* wq = Wq + (size_t)l * DIM * INNER;
        const float* wk = Wk + (size_t)l * DIM * INNER;
        const float* wv = Wv + (size_t)l * DIM * INNER;
        const float* wo = Wo + (size_t)l * INNER * DIM;
        const float* pj = proj + (size_t)l * MFEAT * DH;

        // h = LN1(x)
        ln_kernel<<<NTOK, 256, 0, stream>>>(x, ln1_g + l * DIM, ln1_b + l * DIM, h);

        // q/k/v = h @ W
        launch_gemm(stream, 0, 1, h, DIM, 1, 0, 1.f, wq, INNER, 1, 0, q, INNER, 1, 0,
                    nullptr, 0, nullptr, 0, 0, 0, nullptr, 0, NTOK, INNER, DIM, 0, 1);
        launch_gemm(stream, 0, 1, h, DIM, 1, 0, 1.f, wk, INNER, 1, 0, k, INNER, 1, 0,
                    nullptr, 0, nullptr, 0, 0, 0, nullptr, 0, NTOK, INNER, DIM, 0, 1);
        launch_gemm(stream, 0, 1, h, DIM, 1, 0, 1.f, wv, INNER, 1, 0, v, INNER, 1, 0,
                    nullptr, 0, nullptr, 0, 0, 0, nullptr, 0, NTOK, INNER, DIM, 0, 1);

        // diag terms
        diag_kernel<<<(HN + 255) / 256, 256, 0, stream>>>(q, diagq);
        diag_kernel<<<(HN + 255) / 256, 256, 0, stream>>>(k, diagk);

        // xd = (q*dnorm) @ proj^T  per head:  A[b]=q[:, h*64:(h+1)*64], B=proj^T
        launch_gemm(stream, 0, 0, q, INNER, 1, DH, dnorm, pj, 1, DH, 0,
                    qf, MFEAT, 1, (long)NTOK * MFEAT,
                    nullptr, 0, nullptr, 0, 0, 0, nullptr, 0, NTOK, MFEAT, DH, 0, HEADS);
        launch_gemm(stream, 0, 0, k, INNER, 1, DH, dnorm, pj, 1, DH, 0,
                    kf, MFEAT, 1, (long)NTOK * MFEAT,
                    nullptr, 0, nullptr, 0, 0, 0, nullptr, 0, NTOK, MFEAT, DH, 0, HEADS);

        // stabilizing maxima
        rowmax_kernel<<<(HN + 255) / 256, 256, 0, stream>>>(qf, mxq);
        rowmax_kernel<<<(HN + 255) / 256, 256, 0, stream>>>(kf, mxrows);
        headmax_kernel<<<HEADS, 256, 0, stream>>>(mxrows, mxk);

        // positive random features
        expq_kernel<<<HN * (MFEAT / 256), 256, 0, stream>>>(qf, diagq, mxq);
        expk_kernel<<<HN * (MFEAT / 256), 256, 0, stream>>>(kf, diagk, mxk, mask);

        // k_sum and d_inv
        colsum_kernel<<<HEADS, MFEAT, 0, stream>>>(kf, ksum);
        dinv_kernel<<<(HN + 255) / 256, 256, 0, stream>>>(qf, ksum, dinv);

        // ctx[h] = kf[h]^T @ v[h]   (M=256, K=4096, N=64); A is m-contiguous
        launch_gemm(stream, 1, 1, kf, 1, MFEAT, (long)NTOK * MFEAT, 1.f,
                    v, INNER, 1, DH,
                    ctx, DH, 1, (long)MFEAT * DH,
                    nullptr, 0, nullptr, 0, 0, 0, nullptr, 0, MFEAT, DH, NTOK, 0, HEADS);

        // o[h] = (qf[h] @ ctx[h]) * d_inv   -> interleaved [tok, h*64+d]
        launch_gemm(stream, 0, 1, qf, MFEAT, 1, (long)NTOK * MFEAT, 1.f,
                    ctx, DH, 1, (long)MFEAT * DH,
                    o, INNER, 1, DH,
                    nullptr, 0, nullptr, 0, 0, 0, dinv, NTOK,
                    NTOK, DH, MFEAT, 0, HEADS);

        // x = x + o @ Wo + bo
        launch_gemm(stream, 0, 1, o, INNER, 1, 0, 1.f, wo, DIM, 1, 0, x, DIM, 1, 0,
                    bo + l * DIM, 0, x, DIM, 1, 0, nullptr, 0,
                    NTOK, DIM, INNER, 0, 1);

        // FFN: h = LN2(x); ff = gelu(h@W1+b1); x = x + ff@W2 + b2
        ln_kernel<<<NTOK, 256, 0, stream>>>(x, ln2_g + l * DIM, ln2_b + l * DIM, h);
        launch_gemm(stream, 0, 1, h, DIM, 1, 0, 1.f, W1 + (size_t)l * DIM * FFD, FFD, 1, 0,
                    kf /* reuse as FF scratch */, FFD, 1, 0,
                    b1 + (size_t)l * FFD, 0, nullptr, 0, 0, 0, nullptr, 0,
                    NTOK, FFD, DIM, 1, 1);
        launch_gemm(stream, 0, 1, kf, FFD, 1, 0, 1.f, W2 + (size_t)l * FFD * DIM, DIM, 1, 0,
                    x, DIM, 1, 0,
                    b2 + l * DIM, 0, x, DIM, 1, 0, nullptr, 0,
                    NTOK, DIM, FFD, 0, 1);
    }

    // final LN + output head (ragged N=102, unaligned ldb -> guarded B path)
    ln_kernel<<<NTOK, 256, 0, stream>>>(x, normf_g, normf_b, h);
    launch_gemm(stream, 0, 2, h, DIM, 1, 0, 1.f, Wout, NOUT, 1, 0,
                (float*)d_out, NOUT, 1, 0,
                bout, 0, nullptr, 0, 0, 0, nullptr, 0,
                NTOK, NOUT, DIM, 0, 1);
}
